// EncoderLayer_42777874268838
// MI455X (gfx1250) — matrix operand order repaired and checked by reading, compiled, not run on hardware
//
#include <hip/hip_runtime.h>
#include <hip/hip_bf16.h>

// ---------------------------------------------------------------------------
// Types for CDNA5 WMMA (wave32): bf16 16x16x32, f32 accumulate
// ---------------------------------------------------------------------------
typedef __attribute__((ext_vector_type(16))) __bf16 v16bf;
typedef __attribute__((ext_vector_type(8)))  float  v8f;

union BFrag {
    v16bf v;
    unsigned int u[8];
    unsigned short s[16];
};

static __device__ __forceinline__ unsigned short f32_to_bf16_bits(float f) {
    unsigned int u = __float_as_uint(f);
    unsigned int r = u + 0x7FFFu + ((u >> 16) & 1u);   // round-to-nearest-even
    return (unsigned short)(r >> 16);
}

#define WMMA_BF16(a, b, c) \
    __builtin_amdgcn_wmma_f32_16x16x32_bf16(false, (a), false, (b), (short)0, (c), false, false)

// Problem constants
#define DM   1024
#define NH   16
#define HD   64
#define FFN_ 4096
#define BB   4
#define SS   1024
#define TOK  (BB * SS)          // 4096
#define QKVN (3 * DM)           // 3072
#define EPSV 1e-12f
#define NEGV (-1e9f)

// ---------------------------------------------------------------------------
// fp32 -> bf16 cast
// ---------------------------------------------------------------------------
__global__ void cast_bf16_kernel(const float* __restrict__ in,
                                 unsigned short* __restrict__ out, int n) {
    int i = blockIdx.x * 256 + threadIdx.x;
    if (i < n) out[i] = f32_to_bf16_bits(in[i]);
}

// ---------------------------------------------------------------------------
// Generic bf16 WMMA GEMM:  C[M,N] = A[M,K] * W[N,K]^T + bias
//   mode 0: store bf16(C)
//   mode 1: store bf16(relu(C))
//   mode 2: store f32 (C + res[M,N])
// Block: 256 threads = 8 waves; macro tile 128(M) x 128(N); wave tile 64x32.
// B (weight) tile staged in LDS via async global->LDS (ASYNCcnt) once per
// block; A fragments loaded directly (they vectorize to global_load_b128).
// ---------------------------------------------------------------------------
__global__ __launch_bounds__(256)
void gemm_bf16_kernel(const unsigned short* __restrict__ A,
                      const unsigned short* __restrict__ W,
                      const float* __restrict__ bias,
                      const float* __restrict__ res,
                      float* __restrict__ outF,
                      unsigned short* __restrict__ outB,
                      int M, int N, int K, int mode) {
    const int lane = threadIdx.x & 31;
    const int wave = threadIdx.x >> 5;
    const int wm   = wave & 1;            // 2 wave rows  (64 rows each)
    const int wn   = wave >> 1;           // 4 wave cols  (32 cols each)
    const int l16  = lane & 15;
    const int hw   = lane >> 4;           // half-wave select
    const int blockM = blockIdx.y * 128;
    const int blockN = blockIdx.x * 128;
    const int tid = threadIdx.x;

    // LDS-staged B tile: [128 n-rows][32 k halves], 64B per row, 8 KiB
    __shared__ unsigned short tileB[128 * 32];
    const unsigned ldsBase = (unsigned)(unsigned long long)(void*)tileB;

    // A-fragment k offsets (16x32 bf16 A layout), per VGPR i
    int kaOff[8];
#pragma unroll
    for (int i = 0; i < 8; ++i)
        kaOff[i] = hw * 16 + 2 * i;

    const int rowA0 = blockM + wm * 64 + l16;          // + r*16

    v8f acc[4][2];
#pragma unroll
    for (int r = 0; r < 4; ++r)
#pragma unroll
        for (int t = 0; t < 2; ++t) acc[r][t] = (v8f){};

    for (int kb = 0; kb < K; kb += 32) {
        // ---- async stage of W tile into LDS (each thread copies 2x16B) ----
#pragma unroll
        for (int j = 0; j < 2; ++j) {
            int id = tid + j * 256;                    // 0..511 chunks of 16B
            int n  = id >> 2;                          // 0..127
            int c  = id & 3;                           // 16B chunk in row
            const unsigned short* src = W + (size_t)(blockN + n) * K + kb + c * 8;
            unsigned ldst = ldsBase + (unsigned)(n * 64 + c * 16);
            __asm__ volatile("global_load_async_to_lds_b128 %0, %1, off"
                             :: "v"(ldst),
                                "v"((unsigned long long)(size_t)src)
                             : "memory");
        }

        // ---- A fragments direct from global ----
        BFrag af[4];
#pragma unroll
        for (int r = 0; r < 4; ++r) {
            const unsigned short* ap = A + (size_t)(rowA0 + r * 16) * K + kb;
#pragma unroll
            for (int i = 0; i < 8; ++i)
                af[r].u[i] = *(const unsigned int*)(ap + kaOff[i]);
        }
        if (kb + 32 < K)                    // speculative prefetch, next K slab
            __builtin_prefetch(A + (size_t)rowA0 * K + kb + 32, 0, 0);

        // ---- drain our async copies, then block-wide barrier ----
        __asm__ volatile("s_wait_asynccnt 0" ::: "memory");
        __syncthreads();

        // ---- B fragments from LDS (32B contiguous -> ds_load_b128) ----
        BFrag bf[2];
#pragma unroll
        for (int t = 0; t < 2; ++t) {
            const unsigned short* bp = &tileB[(wn * 32 + t * 16 + l16) * 32 + hw * 16];
#pragma unroll
            for (int i = 0; i < 8; ++i)
                bf[t].u[i] = *(const unsigned int*)(bp + 2 * i);
        }

#pragma unroll
        for (int r = 0; r < 4; ++r)
#pragma unroll
            for (int t = 0; t < 2; ++t)
                acc[r][t] = WMMA_BF16(af[r].v, bf[t].v, acc[r][t]);

        __syncthreads();   // protect tileB before next slab's async writes
    }

    // Epilogue (C layout: vgpr i -> row m = i + 8*hw, col n = l16)
#pragma unroll
    for (int r = 0; r < 4; ++r) {
#pragma unroll
        for (int t = 0; t < 2; ++t) {
#pragma unroll
            for (int i = 0; i < 8; ++i) {
                int row = blockM + wm * 64 + r * 16 + i + 8 * hw;
                int col = blockN + wn * 32 + t * 16 + l16;
                float vv = acc[r][t][i] + bias[col];
                if (mode == 1) vv = fmaxf(vv, 0.0f);
                size_t idx = (size_t)row * N + col;
                if (mode == 2) {
                    outF[idx] = vv + res[idx];
                } else {
                    outB[idx] = f32_to_bf16_bits(vv);
                }
            }
        }
    }
}

// ---------------------------------------------------------------------------
// Flash attention with key masking.
//   qkv: bf16 [TOK, 3072] (q | k | v), ctx: bf16 [TOK, 1024]
// Block: 256 threads = 8 waves; each wave owns 16 queries of one (b,h).
// Grid: (S/128, B*H)
// ---------------------------------------------------------------------------
__global__ __launch_bounds__(256)
void attn_kernel(const unsigned short* __restrict__ qkv,
                 const int* __restrict__ mask,
                 unsigned short* __restrict__ ctx) {
    const int lane = threadIdx.x & 31;
    const int wave = threadIdx.x >> 5;
    const int l16  = lane & 15;
    const int hw   = lane >> 4;
    const int bh   = blockIdx.y;
    const int b    = bh >> 4;           // / NH
    const int h    = bh & 15;           // % NH
    const int q0   = blockIdx.x * 128 + wave * 16;
    const size_t tokbase = (size_t)b * SS;
    const float  scale = 0.125f;        // hd^-0.5

    __shared__ unsigned short pbuf[8][16][34];   // per-wave P staging (pad 34)

    // Load Q fragments (A layout, K=64 -> two 16x32 chunks)
    BFrag qf[2];
    {
        const unsigned short* qp =
            qkv + (size_t)(tokbase + q0 + l16) * QKVN + h * HD;
#pragma unroll
        for (int c = 0; c < 2; ++c)
#pragma unroll
            for (int i = 0; i < 8; ++i) {
                int kd = c * 32 + hw * 16 + 2 * i;
                qf[c].u[i] = *(const unsigned int*)(qp + kd);
            }
    }

    v8f acc[4];
#pragma unroll
    for (int t = 0; t < 4; ++t) acc[t] = (v8f){};
    float mstate[8], lstate[8];
#pragma unroll
    for (int i = 0; i < 8; ++i) { mstate[i] = -3.0e38f; lstate[i] = 0.0f; }

    const int* mrow = mask + (size_t)bh * SS;

    for (int kb = 0; kb < SS; kb += 32) {
        v8f sc[2];
        // ---- scores: Q (16x64) . K^T (64x32), two 16-key column tiles ----
#pragma unroll
        for (int nt = 0; nt < 2; ++nt) {
            int key = kb + nt * 16 + l16;
            const unsigned short* kp =
                qkv + (size_t)(tokbase + key) * QKVN + DM + h * HD;
            BFrag kf[2];
#pragma unroll
            for (int c = 0; c < 2; ++c)
#pragma unroll
                for (int i = 0; i < 8; ++i) {
                    int kd = c * 32 + hw * 16 + 2 * i;
                    kf[c].u[i] = *(const unsigned int*)(kp + kd);
                }
            v8f z = (v8f){};
            z = WMMA_BF16(qf[0].v, kf[0].v, z);
            z = WMMA_BF16(qf[1].v, kf[1].v, z);
            float keep = (mrow[key] != 0) ? 1.0f : 0.0f;
#pragma unroll
            for (int i = 0; i < 8; ++i)
                sc[nt][i] = (keep != 0.0f) ? z[i] * scale : NEGV;
        }

        // ---- online softmax, rows live per (vgpr i, half-wave) ----
#pragma unroll
        for (int i = 0; i < 8; ++i) {
            float mx = fmaxf(sc[0][i], sc[1][i]);
#pragma unroll
            for (int d = 1; d < 16; d <<= 1)
                mx = fmaxf(mx, __shfl_xor(mx, d, 32));
            float mnew = fmaxf(mstate[i], mx);
            float corr = __expf(mstate[i] - mnew);
            float p0 = __expf(sc[0][i] - mnew);
            float p1 = __expf(sc[1][i] - mnew);
            float rs = p0 + p1;
#pragma unroll
            for (int d = 1; d < 16; d <<= 1)
                rs += __shfl_xor(rs, d, 32);
            lstate[i] = lstate[i] * corr + rs;
            mstate[i] = mnew;
#pragma unroll
            for (int t = 0; t < 4; ++t) acc[t][i] *= corr;
            int row = i + 8 * hw;
            pbuf[wave][row][l16]      = f32_to_bf16_bits(p0);
            pbuf[wave][row][16 + l16] = f32_to_bf16_bits(p1);
        }
        // same-wave LDS RAW: CDNA5 split counter wait
        __asm__ volatile("s_wait_dscnt 0" ::: "memory");

        // ---- reload P in A layout (C layout -> A layout via LDS) ----
        BFrag pf;
#pragma unroll
        for (int i = 0; i < 8; ++i) {
            int kk = hw * 16 + 2 * i;
            pf.u[i] = *(const unsigned int*)&pbuf[wave][l16][kk];
        }

        // ---- ctx += P (16x32) . V (32x64), four 16-col output tiles ----
#pragma unroll
        for (int t = 0; t < 4; ++t) {
            BFrag vf;
#pragma unroll
            for (int i = 0; i < 8; ++i) {
                int kd = hw * 16 + 2 * i;     // key within chunk
                const unsigned short* vp =
                    qkv + (size_t)(tokbase + kb + kd) * QKVN + 2 * DM + h * HD +
                    t * 16 + l16;
                unsigned int lo = vp[0];
                unsigned int hi = vp[QKVN];   // key kd+1
                vf.u[i] = lo | (hi << 16);
            }
            acc[t] = WMMA_BF16(pf.v, vf.v, acc[t]);
        }
    }

    // ---- normalize and store ctx (bf16) ----
#pragma unroll
    for (int i = 0; i < 8; ++i) {
        float inv = 1.0f / lstate[i];
        int row = q0 + i + 8 * hw;
#pragma unroll
        for (int t = 0; t < 4; ++t) {
            int col = h * HD + t * 16 + l16;
            ctx[(tokbase + row) * (size_t)DM + col] =
                f32_to_bf16_bits(acc[t][i] * inv);
        }
    }
}

// ---------------------------------------------------------------------------
// LayerNorm over rows of [TOK, 1024]; input already contains residual.
// Writes f32 out and optionally a bf16 copy for the next GEMM.
// ---------------------------------------------------------------------------
__global__ __launch_bounds__(256)
void ln_kernel(const float* __restrict__ in, const float* __restrict__ g,
               const float* __restrict__ be, float* __restrict__ outF,
               unsigned short* __restrict__ outB) {
    const int row  = blockIdx.x;
    const int lane = threadIdx.x & 31;
    const int wave = threadIdx.x >> 5;
    const float* x = in + (size_t)row * DM;

    float v[4];
#pragma unroll
    for (int j = 0; j < 4; ++j) v[j] = x[threadIdx.x + j * 256];

    __shared__ float red[8];

    float s = v[0] + v[1] + v[2] + v[3];
#pragma unroll
    for (int d = 1; d < 32; d <<= 1) s += __shfl_xor(s, d, 32);
    if (lane == 0) red[wave] = s;
    __syncthreads();
    float tot = 0.0f;
#pragma unroll
    for (int w = 0; w < 8; ++w) tot += red[w];
    const float mean = tot * (1.0f / DM);
    __syncthreads();

    float vs = 0.0f;
#pragma unroll
    for (int j = 0; j < 4; ++j) {
        float d0 = v[j] - mean;
        vs += d0 * d0;
    }
#pragma unroll
    for (int d = 1; d < 32; d <<= 1) vs += __shfl_xor(vs, d, 32);
    if (lane == 0) red[wave] = vs;
    __syncthreads();
    float vtot = 0.0f;
#pragma unroll
    for (int w = 0; w < 8; ++w) vtot += red[w];
    const float rstd = rsqrtf(vtot * (1.0f / DM) + EPSV);

#pragma unroll
    for (int j = 0; j < 4; ++j) {
        int col = threadIdx.x + j * 256;
        float y = g[col] * (v[j] - mean) * rstd + be[col];
        size_t idx = (size_t)row * DM + col;
        outF[idx] = y;
        if (outB) outB[idx] = f32_to_bf16_bits(y);
    }
}

// ---------------------------------------------------------------------------
// Host-side orchestration
// ---------------------------------------------------------------------------
extern "C" void kernel_launch(void* const* d_in, const int* in_sizes, int n_in,
                              void* d_out, int out_size, void* d_ws, size_t ws_size,
                              hipStream_t stream) {
    (void)in_sizes; (void)n_in; (void)out_size; (void)ws_size;

    const float* x    = (const float*)d_in[0];
    const int*   mask = (const int*)d_in[1];
    const float* Wqkv = (const float*)d_in[2];
    const float* bqkv = (const float*)d_in[3];
    const float* Wo   = (const float*)d_in[4];
    const float* bo   = (const float*)d_in[5];
    const float* W1   = (const float*)d_in[6];
    const float* b1   = (const float*)d_in[7];
    const float* W2   = (const float*)d_in[8];
    const float* b2   = (const float*)d_in[9];
    const float* g1   = (const float*)d_in[10];
    const float* be1  = (const float*)d_in[11];
    const float* g2   = (const float*)d_in[12];
    const float* be2  = (const float*)d_in[13];
    float* out = (float*)d_out;

    char* ws = (char*)d_ws;
    const size_t MiB = 1024 * 1024;
    unsigned short* xb     = (unsigned short*)(ws +   0 * MiB);  //  8 MiB
    unsigned short* wqkv_b = (unsigned short*)(ws +   8 * MiB);  //  6 MiB
    unsigned short* wo_b   = (unsigned short*)(ws +  14 * MiB);  //  2 MiB
    unsigned short* w1_b   = (unsigned short*)(ws +  16 * MiB);  //  8 MiB
    unsigned short* w2_b   = (unsigned short*)(ws +  24 * MiB);  //  8 MiB
    unsigned short* qkv_b  = (unsigned short*)(ws +  32 * MiB);  // 24 MiB
    unsigned short* ctx_b  = (unsigned short*)(ws +  56 * MiB);  //  8 MiB
    float*          tmp1   = (float*)         (ws +  64 * MiB);  // 16 MiB
    float*          hbuf   = (float*)         (ws +  80 * MiB);  // 16 MiB
    unsigned short* hb     = (unsigned short*)(ws +  96 * MiB);  //  8 MiB
    unsigned short* ff1_b  = (unsigned short*)(ws + 104 * MiB);  // 32 MiB
    float*          tmp2   = (float*)         (ws + 136 * MiB);  // 16 MiB

    // 1) casts to bf16
    auto cast = [&](const float* src, unsigned short* dst, int n) {
        cast_bf16_kernel<<<(n + 255) / 256, 256, 0, stream>>>(src, dst, n);
    };
    cast(x,    xb,     TOK * DM);
    cast(Wqkv, wqkv_b, QKVN * DM);
    cast(Wo,   wo_b,   DM * DM);
    cast(W1,   w1_b,   FFN_ * DM);
    cast(W2,   w2_b,   DM * FFN_);

    // 2) QKV projection: [TOK, 3072] = xb * Wqkv^T + bqkv  -> bf16
    gemm_bf16_kernel<<<dim3(QKVN / 128, TOK / 128), 256, 0, stream>>>(
        xb, wqkv_b, bqkv, nullptr, nullptr, qkv_b, TOK, QKVN, DM, 0);

    // 3) masked flash attention -> ctx bf16
    attn_kernel<<<dim3(SS / 128, BB * NH), 256, 0, stream>>>(qkv_b, mask, ctx_b);

    // 4) out projection + residual(x) -> tmp1 f32
    gemm_bf16_kernel<<<dim3(DM / 128, TOK / 128), 256, 0, stream>>>(
        ctx_b, wo_b, bo, x, tmp1, nullptr, TOK, DM, DM, 2);

    // 5) LayerNorm1 -> h (f32) + hb (bf16)
    ln_kernel<<<TOK, 256, 0, stream>>>(tmp1, g1, be1, hbuf, hb);

    // 6) FFN first GEMM + ReLU -> ff1 bf16
    gemm_bf16_kernel<<<dim3(FFN_ / 128, TOK / 128), 256, 0, stream>>>(
        hb, w1_b, b1, nullptr, nullptr, ff1_b, TOK, FFN_, DM, 1);

    // 7) FFN second GEMM + residual(h) -> tmp2 f32
    gemm_bf16_kernel<<<dim3(DM / 128, TOK / 128), 256, 0, stream>>>(
        ff1_b, w2_b, b2, hbuf, tmp2, nullptr, TOK, DM, FFN_, 2);

    // 8) LayerNorm2 -> final output f32
    ln_kernel<<<TOK, 256, 0, stream>>>(tmp2, g2, be2, out, nullptr);
}
